// AttentionHead_16243566313953
// MI455X (gfx1250) — matrix operand order repaired and checked by reading, not compiled, not run on hardware
//
#include <hip/hip_runtime.h>
#include <hip/hip_bf16.h>

typedef __attribute__((ext_vector_type(16))) _Float16 v16h;
typedef __attribute__((ext_vector_type(8)))  _Float16 v8h;
typedef __attribute__((ext_vector_type(8)))  float    v8f;

#define B_  8
#define T_  2048
#define C_  1024
#define D_  128

union AV { v16h v; v8h h[2]; };

__device__ __forceinline__ v8f wmma16(v16h a, v16h b, v8f c) {
  // D = A(16x32 f16) * B(32x16 f16) + C(16x16 f32)
  return __builtin_amdgcn_wmma_f32_16x16x32_f16(false, a, false, b, (short)0, c,
                                                false, false);
}

// A-operand loader: 16x32 tile from row-major memory (row length = stride halves).
// lane<16: row=lane, K chunks [kbase,kbase+8) and [kbase+16,kbase+24)
// lane>=16: same row, chunks shifted by +8 (per CDNA5 ISA A-layout table).
__device__ __forceinline__ v16h load_a(const _Float16* base, int row, int kbase,
                                       int stride, int lane) {
  const _Float16* p = base + (size_t)row * stride + kbase + ((lane & 16) ? 8 : 0);
  AV a;
  a.h[0] = *(const v8h*)p;
  a.h[1] = *(const v8h*)(p + 16);
  return a.v;
}

// B-operand loader: 32x16 tile read from N-major memory (rows are columns of B,
// K contiguous). lane<16 holds K [kbase,kbase+8),[kbase+16,+24); lane>=16 shifted by +8.
__device__ __forceinline__ v16h load_b(const _Float16* base, int nrow, int kbase,
                                       int stride, int lane) {
  const _Float16* p = base + (size_t)nrow * stride + kbase + ((lane & 16) ? 8 : 0);
  AV b;
  b.h[0] = *(const v8h*)p;
  b.h[1] = *(const v8h*)(p + 16);
  return b.v;
}

__device__ __forceinline__ float redmax16(float v) {
  v = fmaxf(v, __shfl_xor(v, 1));
  v = fmaxf(v, __shfl_xor(v, 2));
  v = fmaxf(v, __shfl_xor(v, 4));
  v = fmaxf(v, __shfl_xor(v, 8));
  return v;
}
__device__ __forceinline__ float redsum16(float v) {
  v += __shfl_xor(v, 1);
  v += __shfl_xor(v, 2);
  v += __shfl_xor(v, 4);
  v += __shfl_xor(v, 8);
  return v;
}

// ---------------------------------------------------------------- convert
__global__ void cvt_f32_to_f16(const float* __restrict__ in,
                               _Float16* __restrict__ out, int n) {
  int i = blockIdx.x * blockDim.x + threadIdx.x;
  int stride = gridDim.x * blockDim.x;
  for (; i < n; i += stride) out[i] = (_Float16)in[i];
}

// ---------------------------------------------------------------- QKV projection
// One wave per 16-row tile of x. Double-buffered k-steps: the A tile + all 8 W
// tiles for step k+1 are issued (distinct registers) before the 8 WMMAs of step
// k, so the per-step s_wait_loadcnt overlaps a full WMMA burst instead of
// serializing load->wait->wmma.
struct KBatch { v16h a; v16h b[8]; };

__device__ __forceinline__ KBatch load_batch(const _Float16* __restrict__ xh,
                                             const _Float16* __restrict__ W,
                                             int rowA, int kk, int lane, int nl) {
  KBatch t;
  t.a = load_a(xh, rowA, kk, C_, lane);
  #pragma unroll
  for (int n = 0; n < 8; ++n) t.b[n] = load_b(W, n * 16 + nl, kk, C_, lane);
  return t;
}

__global__ __launch_bounds__(32)
void qkv_kernel(const _Float16* __restrict__ xh,   // [B*T, C]
                const _Float16* __restrict__ wh,   // [3][D, C]
                const float* __restrict__ bq, const float* __restrict__ bk,
                const float* __restrict__ bv,
                _Float16* __restrict__ qh, _Float16* __restrict__ kh,
                _Float16* __restrict__ vT) {
  const int lane = threadIdx.x;
  const int nl   = lane & 15;
  const int rhi  = (lane & 16) ? 8 : 0;
  const int rt   = blockIdx.x;           // 16-row tile over B*T
  const int rowA = rt * 16 + nl;
  const v8f z = {};

  for (int p = 0; p < 3; ++p) {
    const _Float16* W = wh + (size_t)p * (D_ * C_);
    const float* bias = (p == 0) ? bq : (p == 1) ? bk : bv;
    v8f acc[8];
    #pragma unroll
    for (int n = 0; n < 8; ++n) acc[n] = z;

    KBatch cur = load_batch(xh, W, rowA, 0, lane, nl);
    #pragma unroll 2
    for (int kk = 0; kk < C_; kk += 32) {
      KBatch nxt;
      if (kk + 32 < C_) nxt = load_batch(xh, W, rowA, kk + 32, lane, nl);
      #pragma unroll
      for (int n = 0; n < 8; ++n) acc[n] = wmma16(cur.a, cur.b[n], acc[n]);
      cur = nxt;
    }

    #pragma unroll
    for (int n = 0; n < 8; ++n) {
      const int d = n * 16 + nl;
      const float bval = bias[d];
      #pragma unroll
      for (int r = 0; r < 8; ++r) {
        const int rowg = rt * 16 + r + rhi;
        const float val = acc[n][r] + bval;
        if (p == 2) {                                   // v -> transposed store
          const int bi = rowg >> 11;                    // / T_
          const int t  = rowg & (T_ - 1);
          vT[((size_t)bi * D_ + d) * T_ + t] = (_Float16)val;
        } else {
          _Float16* o = (p == 0) ? qh : kh;
          o[(size_t)rowg * D_ + d] = (_Float16)val;
        }
      }
    }
  }
}

// ---------------------------------------------------------------- flash attention
// One wave per (batch, 16-query tile). Streams 32-key blocks with online softmax.
// Per block: V tiles are issued FIRST (loads retire in order, so the wait before
// the S-WMMAs also completes V; softmax VALU then overlaps nothing but ready
// registers feed P*V immediately). K tiles get 1-deep lookahead across d-chunks.
__global__ __launch_bounds__(32)
void attn_kernel(const _Float16* __restrict__ qh, const _Float16* __restrict__ kh,
                 const _Float16* __restrict__ vT, float* __restrict__ out) {
  __shared__ __align__(32) _Float16 pbuf[16 * 32];   // P tile C/D->A relayout
  const int lane = threadIdx.x;
  const int nl   = lane & 15;
  const int rhi  = (lane & 16) ? 8 : 0;
  const int bi   = blockIdx.x >> 7;     // T_/16 = 128 tiles per batch
  const int qt   = blockIdx.x & 127;
  const _Float16* qb = qh + (size_t)bi * T_ * D_;
  const _Float16* kb = kh + (size_t)bi * T_ * D_;
  const _Float16* vb = vT + (size_t)bi * D_ * T_;
  const v8f z = {};

  v16h qa[4];                            // Q tile 16x128 held in registers
  #pragma unroll
  for (int c = 0; c < 4; ++c) qa[c] = load_a(qb, qt * 16 + nl, c * 32, D_, lane);

  v8f acc[8];                            // O accumulator 16x128 f32
  float m[8], l[8], corr[8];
  #pragma unroll
  for (int n = 0; n < 8; ++n) acc[n] = z;
  #pragma unroll
  for (int r = 0; r < 8; ++r) { m[r] = -1e30f; l[r] = 0.0f; }

  const int kbmax = ((qt + 1) * 16 - 1) >> 5;   // last causal 32-key block (incl.)
  for (int kblk = 0; kblk <= kbmax; ++kblk) {
    // ---- issue V tiles early (independent of S; retired by the S wait)
    v16h vt[8];
    #pragma unroll
    for (int n = 0; n < 8; ++n)
      vt[n] = load_b(vb, n * 16 + nl, kblk * 32, T_, lane);

    // ---- S = Q * K^T (16x32 scores), K loads 1-deep pipelined over d-chunks
    v8f s0 = z, s1 = z;
    v16h k0 = load_b(kb, kblk * 32 + nl,      0, D_, lane);
    v16h k1 = load_b(kb, kblk * 32 + 16 + nl, 0, D_, lane);
    #pragma unroll
    for (int c = 0; c < 4; ++c) {
      v16h k0n, k1n;
      if (c < 3) {
        k0n = load_b(kb, kblk * 32 + nl,      (c + 1) * 32, D_, lane);
        k1n = load_b(kb, kblk * 32 + 16 + nl, (c + 1) * 32, D_, lane);
      }
      s0 = wmma16(qa[c], k0, s0);
      s1 = wmma16(qa[c], k1, s1);
      k0 = k0n; k1 = k1n;
    }
    if (kblk < kbmax)   // gfx1250 global_prefetch of next K block
      __builtin_prefetch(kb + (size_t)(kblk + 1) * 32 * D_ + (lane << 4), 0, 1);

    // ---- scale (1/sqrt(C)=1/32), causal mask, online softmax update
    const int col0 = kblk * 32 + nl;
    const int col1 = col0 + 16;
    #pragma unroll
    for (int r = 0; r < 8; ++r) {
      const int row = qt * 16 + r + rhi;
      float v0 = (col0 <= row) ? s0[r] * 0.03125f : -1e30f;
      float v1 = (col1 <= row) ? s1[r] * 0.03125f : -1e30f;
      float mx = redmax16(fmaxf(v0, v1));
      float mn = fmaxf(m[r], mx);
      float cr = __expf(m[r] - mn);
      float p0 = __expf(v0 - mn);
      float p1 = __expf(v1 - mn);
      l[r]   = l[r] * cr + redsum16(p0 + p1);
      m[r]   = mn;
      corr[r] = cr;
      pbuf[(r + rhi) * 32 + nl]      = (_Float16)p0;   // C/D layout -> LDS
      pbuf[(r + rhi) * 32 + 16 + nl] = (_Float16)p1;
    }
    #pragma unroll
    for (int n = 0; n < 8; ++n)
      #pragma unroll
      for (int r = 0; r < 8; ++r) acc[n][r] *= corr[r];

    __syncthreads();                     // single-wave WG: S_NOP + DS ordering
    v16h pa = load_a(pbuf, nl, 0, 32, lane);   // reload P in A layout

    // ---- O += P * V (V tiles already resident in registers), 8 WMMAs
    #pragma unroll
    for (int n = 0; n < 8; ++n) acc[n] = wmma16(pa, vt[n], acc[n]);
    __syncthreads();
  }

  // ---- normalize and store fp32 output [B,T,D]
  #pragma unroll
  for (int r = 0; r < 8; ++r) {
    const float inv = 1.0f / l[r];
    const int row = qt * 16 + r + rhi;
    #pragma unroll
    for (int n = 0; n < 8; ++n)
      out[((size_t)bi * T_ + row) * D_ + n * 16 + nl] = acc[n][r] * inv;
  }
}

// ---------------------------------------------------------------- host
extern "C" void kernel_launch(void* const* d_in, const int* in_sizes, int n_in,
                              void* d_out, int out_size, void* d_ws, size_t ws_size,
                              hipStream_t stream) {
  const float* x  = (const float*)d_in[0];
  const float* Wq = (const float*)d_in[1];
  const float* bq = (const float*)d_in[2];
  const float* Wk = (const float*)d_in[3];
  const float* bk = (const float*)d_in[4];
  const float* Wv = (const float*)d_in[5];
  const float* bv = (const float*)d_in[6];
  float* out = (float*)d_out;

  char* ws = (char*)d_ws;
  const size_t XH_OFF = 0;                         // x f16: 32 MiB
  const size_t WH_OFF = (size_t)32 << 20;          // W f16: 0.75 MiB
  const size_t QH_OFF = (size_t)33 << 20;          // q f16: 4 MiB
  const size_t KH_OFF = (size_t)37 << 20;          // k f16: 4 MiB
  const size_t VT_OFF = (size_t)41 << 20;          // vT f16: 4 MiB
  _Float16* xh = (_Float16*)(ws + XH_OFF);
  _Float16* wh = (_Float16*)(ws + WH_OFF);
  _Float16* qh = (_Float16*)(ws + QH_OFF);
  _Float16* kh = (_Float16*)(ws + KH_OFF);
  _Float16* vT = (_Float16*)(ws + VT_OFF);

  const int BT = B_ * T_;          // 16384
  const int DC = D_ * C_;          // 131072

  cvt_f32_to_f16<<<8192, 256, 0, stream>>>(x, xh, BT * C_);
  cvt_f32_to_f16<<<256, 256, 0, stream>>>(Wq, wh + 0 * (size_t)DC, DC);
  cvt_f32_to_f16<<<256, 256, 0, stream>>>(Wk, wh + 1 * (size_t)DC, DC);
  cvt_f32_to_f16<<<256, 256, 0, stream>>>(Wv, wh + 2 * (size_t)DC, DC);

  qkv_kernel<<<BT / 16, 32, 0, stream>>>(xh, wh, bq, bk, bv, qh, kh, vT);
  attn_kernel<<<BT / 16, 32, 0, stream>>>(qh, kh, vT, out);
  (void)in_sizes; (void)n_in; (void)out_size; (void)ws_size;
}
